// SwinTransformerBlock_11149735100575
// MI455X (gfx1250) — matrix operand order repaired and checked by reading, not compiled, not run on hardware
//
#include <hip/hip_runtime.h>
#include <hip/hip_bf16.h>
#include <math.h>

// ---------------- constants ----------------
#define B_     32
#define H_     52
#define W_     56
#define C_     384
#define WS_    7
#define SHIFT_ 3
#define NH_    12
#define HP_    56
#define HD_    32
#define N_     49
#define NWX_   8
#define NW_    64
#define NWIN_  2048            // B_*NW_
#define TOK_   100352          // NWIN_*N_
#define DFF_   1536
#define HW_    2912            // H_*W_
#define MTOK_  93184           // B_*HW_
#define SCALE_C 0.17677669529663687f
#define NEGBIG (-1e30f)

typedef __attribute__((ext_vector_type(16))) __bf16 v16bf;
typedef __attribute__((ext_vector_type(8)))  float  v8f;

union FragB16 { v16bf v; uint4 u[2]; };
union FragF32 { v8f v; float f[8]; };

__device__ __forceinline__ unsigned short f2bf(float f) {
    union { float f; unsigned u; } v; v.f = f;
    unsigned r = v.u + 0x7fffu + ((v.u >> 16) & 1u);
    return (unsigned short)(r >> 16);
}
__device__ __forceinline__ float bf2f(unsigned short s) {
    union { unsigned u; float f; } v; v.u = ((unsigned)s) << 16;
    return v.f;
}

// ---------------- K0: weight convert + transpose  Wt[n][k] = bf16(W[k][n]) ----
__global__ __launch_bounds__(256)
void transpose_bf16_kernel(const float* __restrict__ w, unsigned short* __restrict__ wt,
                           int K, int N) {
    int i = blockIdx.x * 256 + threadIdx.x;
    if (i >= N * K) return;
    int n = i / K, k = i % K;
    wt[i] = f2bf(w[(size_t)k * N + n]);
}

// ------- K_mask: padded cmbp[w][h][64][64] = pad+attn+rpb, fill = -1e30 -------
__global__ __launch_bounds__(256)
void cmb_kernel(const float* __restrict__ amask, const float* __restrict__ pmask,
                const float* __restrict__ rpb, const int* __restrict__ relidx,
                float* __restrict__ cmbp) {
    int i = blockIdx.x * 256 + threadIdx.x;
    const int TOTAL = NW_ * NH_ * 64 * 64;
    if (i >= TOTAL) return;
    int j  = i & 63;
    int ii = (i >> 6) & 63;
    int h  = (i >> 12) % NH_;
    int w  = i / (4096 * NH_);
    float v = NEGBIG;
    if (ii < N_ && j < N_) {
        int mo = w * (N_ * N_) + ii * N_ + j;
        v = pmask[mo] + amask[mo] + rpb[relidx[ii * N_ + j] * NH_ + h];
    }
    cmbp[i] = v;
}

// ---------------- K1: LN1 + roll + pad + window partition -> bf16 ----------
__global__ __launch_bounds__(128)
void ln1_window_kernel(const float* __restrict__ x, const float* __restrict__ g,
                       const float* __restrict__ bta, unsigned short* __restrict__ win_bf) {
    const int tok = blockIdx.x;           // [0, TOK_)
    const int tid = threadIdx.x;
    const int win = tok / N_, idx = tok % N_;
    const int b   = win >> 6, wloc = win & 63;
    const int wy  = wloc >> 3, wx = wloc & 7;
    const int rr  = idx / WS_, cc = idx % WS_;
    const int hp  = wy * WS_ + rr;
    const int wc  = wx * WS_ + cc;
    unsigned short* out = win_bf + (size_t)tok * C_;
    if (hp < 4) {                          // top padding rows -> zeros (pre-bias)
        for (int c = tid; c < C_; c += 128) out[c] = 0;
        return;
    }
    const int hs  = (hp - 4 + SHIFT_) % H_;
    const int wsr = (wc + SHIFT_) % W_;
    const float* src = x + ((size_t)b * HW_ + hs * W_ + wsr) * C_;
    float v0 = src[tid], v1 = src[tid + 128], v2 = src[tid + 256];
    __shared__ float red[128];
    red[tid] = v0 + v1 + v2;
    __syncthreads();
    #pragma unroll
    for (int o = 64; o > 0; o >>= 1) { if (tid < o) red[tid] += red[tid + o]; __syncthreads(); }
    float mean = red[0] * (1.0f / C_);
    __syncthreads();
    float d0 = v0 - mean, d1 = v1 - mean, d2 = v2 - mean;
    red[tid] = d0 * d0 + d1 * d1 + d2 * d2;
    __syncthreads();
    #pragma unroll
    for (int o = 64; o > 0; o >>= 1) { if (tid < o) red[tid] += red[tid + o]; __syncthreads(); }
    float rstd = rsqrtf(red[0] * (1.0f / C_) + 1e-5f);
    out[tid]       = f2bf(d0 * rstd * g[tid]       + bta[tid]);
    out[tid + 128] = f2bf(d1 * rstd * g[tid + 128] + bta[tid + 128]);
    out[tid + 256] = f2bf(d2 * rstd * g[tid + 256] + bta[tid + 256]);
}

// ---------------- generic bf16 WMMA GEMM: C = A[M,K] @ Bt[N,K]^T + bias ------
// epilogues: 0 = store bf16, 1 = exact GELU then bf16, 2 = fp32 += (residual)
enum { EPI_STORE_BF16 = 0, EPI_GELU_BF16 = 1, EPI_ADD_F32 = 2 };

template <int EPI>
__global__ __launch_bounds__(256)
void gemm_bf16_wmma(const unsigned short* __restrict__ A,
                    const unsigned short* __restrict__ Bt,
                    const float* __restrict__ bias,
                    void* __restrict__ Cout,
                    int M, int N, int K) {
    __shared__ __align__(16) unsigned short As[128 * 32];
    __shared__ __align__(16) unsigned short Bs[128 * 32];
    const int tid  = threadIdx.x;
    const int lane = tid & 31;
    const int wave = tid >> 5;
    const int wm   = wave & 3;   // M sub-tile: wm*32
    const int wn   = wave >> 2;  // N sub-tile: wn*64
    const int m0   = blockIdx.y * 128;
    const int n0   = blockIdx.x * 128;
    const int lrow = tid >> 2;   // 0..63
    const int lq   = tid & 3;    // 16B quarter of a 64B row
    const int lj   = lane & 15;

    v8f zero = {};
    FragF32 acc[2][4];
    #pragma unroll
    for (int i = 0; i < 2; i++)
        #pragma unroll
        for (int j = 0; j < 4; j++) acc[i][j].v = zero;

    for (int k0 = 0; k0 < K; k0 += 32) {
        // stage 128x32 bf16 tiles of A and Bt in LDS (2 x b128 per thread each)
        #pragma unroll
        for (int p = 0; p < 2; p++) {
            int row = lrow + p * 64;
            uint4 av = *(const uint4*)(A  + (size_t)(m0 + row) * K + k0 + lq * 8);
            *(uint4*)(&As[row * 32 + lq * 8]) = av;
            uint4 bv = *(const uint4*)(Bt + (size_t)(n0 + row) * K + k0 + lq * 8);
            *(uint4*)(&Bs[row * 32 + lq * 8]) = bv;
        }
        if (k0 + 32 < K) {  // gfx1250 global_prefetch_b8 of next K tile
            __builtin_prefetch((const void*)(A  + (size_t)(m0 + lrow) * K + k0 + 32), 0, 1);
            __builtin_prefetch((const void*)(Bt + (size_t)(n0 + lrow) * K + k0 + 32), 0, 1);
        }
        __syncthreads();

        FragB16 af[2], bf[4];
        #pragma unroll
        for (int mt = 0; mt < 2; mt++) {
            int row = wm * 32 + mt * 16 + lj;
            int kb  = (lane < 16) ? 0 : 8;           // A frag: K 0-7/16-23 vs 8-15/24-31
            af[mt].u[0] = *(const uint4*)(&As[row * 32 + kb]);
            af[mt].u[1] = *(const uint4*)(&As[row * 32 + kb + 16]);
        }
        #pragma unroll
        for (int nt = 0; nt < 4; nt++) {
            int col = wn * 64 + nt * 16 + lj;
            int kb  = (lane < 16) ? 0 : 8;           // B frag: K 0-7/16-23 vs 8-15/24-31
            bf[nt].u[0] = *(const uint4*)(&Bs[col * 32 + kb]);
            bf[nt].u[1] = *(const uint4*)(&Bs[col * 32 + kb + 16]);
        }
        #pragma unroll
        for (int mt = 0; mt < 2; mt++)
            #pragma unroll
            for (int nt = 0; nt < 4; nt++)
                acc[mt][nt].v = __builtin_amdgcn_wmma_f32_16x16x32_bf16(
                    false, af[mt].v, false, bf[nt].v, (short)0, acc[mt][nt].v, false, false);
        __syncthreads();
    }

    const int lh = lane >> 4;
    #pragma unroll
    for (int mt = 0; mt < 2; mt++) {
        #pragma unroll
        for (int nt = 0; nt < 4; nt++) {
            #pragma unroll
            for (int r = 0; r < 8; r++) {
                int row = m0 + wm * 32 + mt * 16 + r + lh * 8;
                int col = n0 + wn * 64 + nt * 16 + lj;
                float v = acc[mt][nt].f[r] + bias[col];
                size_t o = (size_t)row * N + col;
                if (EPI == EPI_STORE_BF16) {
                    ((unsigned short*)Cout)[o] = f2bf(v);
                } else if (EPI == EPI_GELU_BF16) {
                    float ge = 0.5f * v * (1.0f + erff(v * 0.70710678118654752f));
                    ((unsigned short*)Cout)[o] = f2bf(ge);
                } else {
                    ((float*)Cout)[o] += v;
                }
            }
        }
    }
}

// -------- K_attn: one block per (window, head); 4 waves = 4 query chunks -----
__global__ __launch_bounds__(128)
void attn_kernel(const unsigned short* __restrict__ qkv,  // [TOK_][1152] bf16
                 const float* __restrict__ cmbp,          // [64][12][64][64], pad=-1e30
                 unsigned short* __restrict__ obf) {      // [TOK_][384] bf16
    __shared__ __align__(16) unsigned short qs[64 * 32];
    __shared__ __align__(16) unsigned short ks[64 * 32];
    __shared__ __align__(16) unsigned short vt[32 * 64];      // v transposed [d][j]
    __shared__ __align__(16) unsigned short ps[4][16 * 64];   // per-wave P chunk

    const int tid  = threadIdx.x;
    const int lane = tid & 31;
    const int mc   = tid >> 5;          // wave id == query chunk (16 rows each)
    const int win  = blockIdx.x / NH_;
    const int hh   = blockIdx.x % NH_;
    const int w64  = win & 63;
    const int lj   = lane & 15;
    const int lh   = lane >> 4;

    // zero staging buffers: padded rows/cols must be EXACT zeros so padded
    // scores become 0*SCALE + (-1e30) and padded P@V K-columns contribute 0
    {
        uint4 z4 = make_uint4(0u, 0u, 0u, 0u);
        #pragma unroll
        for (int t = tid; t < (64 * 32) / 8; t += 128) {
            *(uint4*)(&qs[t * 8]) = z4;
            *(uint4*)(&ks[t * 8]) = z4;
            *(uint4*)(&vt[t * 8]) = z4;
        }
    }
    __syncthreads();

    if (tid < N_) {
        const int r = tid;
        size_t base = (size_t)(win * N_ + r) * (3 * C_) + hh * HD_;
        const uint4* qg = (const uint4*)(qkv + base);
        const uint4* kg = (const uint4*)(qkv + base + C_);
        const uint4* vg = (const uint4*)(qkv + base + 2 * C_);
        #pragma unroll
        for (int p = 0; p < 4; p++) *(uint4*)(&qs[r * 32 + p * 8]) = qg[p];
        #pragma unroll
        for (int p = 0; p < 4; p++) *(uint4*)(&ks[r * 32 + p * 8]) = kg[p];
        union { uint4 u[4]; unsigned short s[32]; } vv;
        #pragma unroll
        for (int p = 0; p < 4; p++) vv.u[p] = vg[p];
        #pragma unroll
        for (int d = 0; d < 32; d++) vt[d * 64 + r] = vv.s[d];  // transpose into LDS
    }
    __syncthreads();

    const float* cmbw = cmbp + (size_t)(w64 * NH_ + hh) * 4096;
    v8f zero = {};

    // ---- this wave's 16 query rows ----
    FragB16 qf;
    {
        int row = mc * 16 + lj;
        int kb  = (lane < 16) ? 0 : 8;
        qf.u[0] = *(const uint4*)(&qs[row * 32 + kb]);
        qf.u[1] = *(const uint4*)(&qs[row * 32 + kb + 16]);
    }
    // S = Q @ K^T (4 column tiles of 16)
    FragF32 S[4];
    #pragma unroll
    for (int nt = 0; nt < 4; nt++) {
        FragB16 kf;
        int col = nt * 16 + lj;
        int kb  = (lane < 16) ? 0 : 8;
        kf.u[0] = *(const uint4*)(&ks[col * 32 + kb]);
        kf.u[1] = *(const uint4*)(&ks[col * 32 + kb + 16]);
        S[nt].v = __builtin_amdgcn_wmma_f32_16x16x32_bf16(
            false, qf.v, false, kf.v, (short)0, zero, false, false);
    }
    // scale + combined bias/mask (unconditional, coalesced; padding = -1e30)
    const int ibase = (mc * 16 + lh * 8) * 64 + lj;   // row i, col j index into 64x64
    #pragma unroll
    for (int nt = 0; nt < 4; nt++)
        #pragma unroll
        for (int r = 0; r < 8; r++)
            S[nt].f[r] = S[nt].f[r] * SCALE_C + cmbw[ibase + r * 64 + nt * 16];
    // row softmax: reduce over 4 tiles + 16 lanes of each half-wave
    #pragma unroll
    for (int r = 0; r < 8; r++) {
        float m = fmaxf(fmaxf(S[0].f[r], S[1].f[r]), fmaxf(S[2].f[r], S[3].f[r]));
        #pragma unroll
        for (int off = 8; off >= 1; off >>= 1) m = fmaxf(m, __shfl_xor(m, off, 32));
        float sum = 0.f;
        #pragma unroll
        for (int nt = 0; nt < 4; nt++) {
            float e = __expf(S[nt].f[r] - m);
            S[nt].f[r] = e;
            sum += e;
        }
        #pragma unroll
        for (int off = 8; off >= 1; off >>= 1) sum += __shfl_xor(sum, off, 32);
        float inv = 1.0f / sum;
        #pragma unroll
        for (int nt = 0; nt < 4; nt++) S[nt].f[r] *= inv;
    }
    // spill this wave's P chunk (16 x 64) to its private LDS slice as bf16
    unsigned short* myps = ps[mc];
    #pragma unroll
    for (int nt = 0; nt < 4; nt++)
        #pragma unroll
        for (int r = 0; r < 8; r++)
            myps[(r + lh * 8) * 64 + nt * 16 + lj] = f2bf(S[nt].f[r]);
    // no barrier needed: each wave reads only its own ps slice (in-wave LDS order)
    // O = P @ V   (K = 64 via two 32-steps; 2 output d-tiles)
    FragF32 O0, O1;
    O0.v = zero; O1.v = zero;
    #pragma unroll
    for (int kx = 0; kx < 2; kx++) {
        FragB16 pf;
        {
            int kb = kx * 32 + ((lane < 16) ? 0 : 8);
            pf.u[0] = *(const uint4*)(&myps[lj * 64 + kb]);
            pf.u[1] = *(const uint4*)(&myps[lj * 64 + kb + 16]);
        }
        FragB16 vf0, vf1;
        {
            int kb = kx * 32 + ((lane < 16) ? 0 : 8);
            vf0.u[0] = *(const uint4*)(&vt[lj * 64 + kb]);
            vf0.u[1] = *(const uint4*)(&vt[lj * 64 + kb + 16]);
            vf1.u[0] = *(const uint4*)(&vt[(16 + lj) * 64 + kb]);
            vf1.u[1] = *(const uint4*)(&vt[(16 + lj) * 64 + kb + 16]);
        }
        O0.v = __builtin_amdgcn_wmma_f32_16x16x32_bf16(false, pf.v, false, vf0.v,
                                                       (short)0, O0.v, false, false);
        O1.v = __builtin_amdgcn_wmma_f32_16x16x32_bf16(false, pf.v, false, vf1.v,
                                                       (short)0, O1.v, false, false);
    }
    #pragma unroll
    for (int r = 0; r < 8; r++) {
        int i = mc * 16 + r + lh * 8;
        if (i < N_) {
            size_t base = (size_t)(win * N_ + i) * C_ + hh * HD_;
            obf[base + lj]      = f2bf(O0.f[r]);
            obf[base + 16 + lj] = f2bf(O1.f[r]);
        }
    }
}

// ---------------- K5: window reverse + roll + residual + LN2 ----------------
__global__ __launch_bounds__(128)
void unwin_ln2_kernel(const float* __restrict__ x, const unsigned short* __restrict__ proj_out,
                      const float* __restrict__ g, const float* __restrict__ bta,
                      float* __restrict__ x1out, unsigned short* __restrict__ tbf) {
    const int tok = blockIdx.x;     // b*HW + hh*W + ww
    const int tid = threadIdx.x;
    const int b   = tok / HW_;
    const int rem = tok % HW_;
    const int hhr = rem / W_, ww = rem % W_;
    const int sh  = (hhr - 2 * SHIFT_ + H_) % H_;   // reference rolls by 2*SHIFT
    const int sw  = (ww  - 2 * SHIFT_ + W_) % W_;
    const int hp  = sh + 4;
    const int swin = b * NW_ + (hp / WS_) * NWX_ + (sw / WS_);
    const int sidx = (hp % WS_) * WS_ + (sw % WS_);
    const float* xs = x + (size_t)tok * C_;
    const unsigned short* po = proj_out + (size_t)(swin * N_ + sidx) * C_;
    float v0 = xs[tid]       + bf2f(po[tid]);
    float v1 = xs[tid + 128] + bf2f(po[tid + 128]);
    float v2 = xs[tid + 256] + bf2f(po[tid + 256]);
    float* xo = x1out + (size_t)tok * C_;
    xo[tid] = v0; xo[tid + 128] = v1; xo[tid + 256] = v2;

    __shared__ float red[128];
    red[tid] = v0 + v1 + v2;
    __syncthreads();
    #pragma unroll
    for (int o = 64; o > 0; o >>= 1) { if (tid < o) red[tid] += red[tid + o]; __syncthreads(); }
    float mean = red[0] * (1.0f / C_);
    __syncthreads();
    float d0 = v0 - mean, d1 = v1 - mean, d2 = v2 - mean;
    red[tid] = d0 * d0 + d1 * d1 + d2 * d2;
    __syncthreads();
    #pragma unroll
    for (int o = 64; o > 0; o >>= 1) { if (tid < o) red[tid] += red[tid + o]; __syncthreads(); }
    float rstd = rsqrtf(red[0] * (1.0f / C_) + 1e-5f);
    unsigned short* out = tbf + (size_t)tok * C_;
    out[tid]       = f2bf(d0 * rstd * g[tid]       + bta[tid]);
    out[tid + 128] = f2bf(d1 * rstd * g[tid + 128] + bta[tid + 128]);
    out[tid + 256] = f2bf(d2 * rstd * g[tid + 256] + bta[tid + 256]);
}

// ---------------- launcher ----------------
extern "C" void kernel_launch(void* const* d_in, const int* in_sizes, int n_in,
                              void* d_out, int out_size, void* d_ws, size_t ws_size,
                              hipStream_t stream) {
    const float* x      = (const float*)d_in[0];
    const float* n1g    = (const float*)d_in[1];
    const float* n1b    = (const float*)d_in[2];
    const float* qkv_w  = (const float*)d_in[3];
    const float* qkv_b  = (const float*)d_in[4];
    const float* rpb    = (const float*)d_in[5];
    const float* proj_w = (const float*)d_in[6];
    const float* proj_b = (const float*)d_in[7];
    const float* n2g    = (const float*)d_in[8];
    const float* n2b    = (const float*)d_in[9];
    const float* fc1_w  = (const float*)d_in[10];
    const float* fc1_b  = (const float*)d_in[11];
    const float* fc2_w  = (const float*)d_in[12];
    const float* fc2_b  = (const float*)d_in[13];
    const float* amask  = (const float*)d_in[14];
    const float* pmask  = (const float*)d_in[15];
    const int*   relidx = (const int*)d_in[16];
    float* out = (float*)d_out;

    char* ws = (char*)d_ws;
    size_t off = 0;
    auto alloc = [&](size_t bytes) -> char* {
        char* p = ws + off;
        off += (bytes + 255) & ~(size_t)255;
        return p;
    };
    // R1 reused sequentially: win_bf -> o_bf -> t_bf (each consumed before next write)
    unsigned short* R1     = (unsigned short*)alloc((size_t)TOK_ * C_ * 2);
    unsigned short* qkvT   = (unsigned short*)alloc((size_t)(3 * C_) * C_ * 2);
    unsigned short* projT  = (unsigned short*)alloc((size_t)C_ * C_ * 2);
    unsigned short* fc1T   = (unsigned short*)alloc((size_t)DFF_ * C_ * 2);
    unsigned short* fc2T   = (unsigned short*)alloc((size_t)C_ * DFF_ * 2);
    // R2 reused: qkv_out -> proj_out
    unsigned short* R2     = (unsigned short*)alloc((size_t)TOK_ * 3 * C_ * 2);
    float*          cmbp   = (float*)alloc((size_t)NW_ * NH_ * 64 * 64 * 4);
    unsigned short* gelu   = (unsigned short*)alloc((size_t)MTOK_ * DFF_ * 2);

    unsigned short* win_bf   = R1;
    unsigned short* o_bf     = R1;
    unsigned short* t_bf     = R1;
    unsigned short* qkv_out  = R2;
    unsigned short* proj_out = R2;

    // weights (bf16, transposed) + combined attention bias (padded 64x64)
    transpose_bf16_kernel<<<((3 * C_) * C_ + 255) / 256, 256, 0, stream>>>(qkv_w, qkvT, C_, 3 * C_);
    transpose_bf16_kernel<<<(C_ * C_ + 255) / 256, 256, 0, stream>>>(proj_w, projT, C_, C_);
    transpose_bf16_kernel<<<(C_ * DFF_ + 255) / 256, 256, 0, stream>>>(fc1_w, fc1T, C_, DFF_);
    transpose_bf16_kernel<<<(DFF_ * C_ + 255) / 256, 256, 0, stream>>>(fc2_w, fc2T, DFF_, C_);
    cmb_kernel<<<(NW_ * NH_ * 64 * 64 + 255) / 256, 256, 0, stream>>>(amask, pmask, rpb, relidx, cmbp);

    // LN1 + shift + pad + window partition
    ln1_window_kernel<<<TOK_, 128, 0, stream>>>(x, n1g, n1b, win_bf);

    // QKV GEMM: [100352 x 384] @ [384 x 1152]
    gemm_bf16_wmma<EPI_STORE_BF16><<<dim3((3 * C_) / 128, TOK_ / 128), 256, 0, stream>>>(
        win_bf, qkvT, qkv_b, qkv_out, TOK_, 3 * C_, C_);

    // attention: one block per (window, head), 4 waves = 4 query chunks
    attn_kernel<<<NWIN_ * NH_, 128, 0, stream>>>(qkv_out, cmbp, o_bf);

    // projection GEMM: [100352 x 384] @ [384 x 384]
    gemm_bf16_wmma<EPI_STORE_BF16><<<dim3(C_ / 128, TOK_ / 128), 256, 0, stream>>>(
        o_bf, projT, proj_b, proj_out, TOK_, C_, C_);

    // window reverse + roll + residual (x1 -> d_out) + LN2 (-> t_bf)
    unwin_ln2_kernel<<<MTOK_, 128, 0, stream>>>(x, proj_out, n2g, n2b, out, t_bf);

    // MLP fc1 + exact GELU: [93184 x 384] @ [384 x 1536]
    gemm_bf16_wmma<EPI_GELU_BF16><<<dim3(DFF_ / 128, MTOK_ / 128), 256, 0, stream>>>(
        t_bf, fc1T, fc1_b, gelu, MTOK_, DFF_, C_);

    // MLP fc2 + residual accumulate into d_out: [93184 x 1536] @ [1536 x 384]
    gemm_bf16_wmma<EPI_ADD_F32><<<dim3(C_ / 128, MTOK_ / 128), 256, 0, stream>>>(
        gelu, fc2T, fc2_b, out, MTOK_, C_, DFF_);
}
